// CrossMaskedMHCA_15728170238330
// MI455X (gfx1250) — matrix operand order repaired and checked by reading, compiled, not run on hardware
//
#include <hip/hip_runtime.h>
#include <hip/hip_bf16.h>
#include <stdint.h>

#define BB 4
#define CC 1024
#define TT 1024
#define HH 16
#define CHD 64
#define LN_EPS 1e-5f

typedef __bf16 bf16_t;
typedef __attribute__((ext_vector_type(16))) __bf16 v16bf;
typedef __attribute__((ext_vector_type(8)))  __bf16 v8bf;
typedef __attribute__((ext_vector_type(4)))  __bf16 v4bf;
typedef __attribute__((ext_vector_type(8)))  float  v8f;

// ---- Fused DPP16 reductions: one VOP2-DPP instruction per step (shuffle+op),
// reducing across a 16-lane group (one C/D-tile row). No LDS, no waits.
__device__ __forceinline__ float dpp_max16(float v) {
  float t;
  asm("v_max_num_f32_dpp %0, %1, %1 quad_perm:[1,0,3,2] row_mask:0xf bank_mask:0xf bound_ctrl:1"
      : "=v"(t) : "v"(v));
  asm("v_max_num_f32_dpp %0, %1, %1 quad_perm:[2,3,0,1] row_mask:0xf bank_mask:0xf bound_ctrl:1"
      : "=v"(v) : "v"(t));
  asm("v_max_num_f32_dpp %0, %1, %1 row_half_mirror row_mask:0xf bank_mask:0xf bound_ctrl:1"
      : "=v"(t) : "v"(v));
  asm("v_max_num_f32_dpp %0, %1, %1 row_mirror row_mask:0xf bank_mask:0xf bound_ctrl:1"
      : "=v"(v) : "v"(t));
  return v;
}
__device__ __forceinline__ float dpp_sum16(float v) {
  float t;
  asm("v_add_f32_dpp %0, %1, %1 quad_perm:[1,0,3,2] row_mask:0xf bank_mask:0xf bound_ctrl:1"
      : "=v"(t) : "v"(v));
  asm("v_add_f32_dpp %0, %1, %1 quad_perm:[2,3,0,1] row_mask:0xf bank_mask:0xf bound_ctrl:1"
      : "=v"(v) : "v"(t));
  asm("v_add_f32_dpp %0, %1, %1 row_half_mirror row_mask:0xf bank_mask:0xf bound_ctrl:1"
      : "=v"(t) : "v"(v));
  asm("v_add_f32_dpp %0, %1, %1 row_mirror row_mask:0xf bank_mask:0xf bound_ctrl:1"
      : "=v"(v) : "v"(t));
  return v;
}

#define SHUF16(lo, hi) \
  __builtin_shufflevector(lo, hi, 0, 1, 2, 3, 4, 5, 6, 7, 8, 9, 10, 11, 12, 13, 14, 15)

// ---------------- Stage 0: one-time f32 -> bf16 weight conversion (row-major kept)
__global__ void cvt_w_kernel(const float* __restrict__ w, bf16_t* __restrict__ o) {
  int i = (blockIdx.x * blockDim.x + threadIdx.x) * 4;
  float4 v = *(const float4*)(w + i);
  v4bf r;
  r[0] = (bf16_t)v.x; r[1] = (bf16_t)v.y; r[2] = (bf16_t)v.z; r[3] = (bf16_t)v.w;
  *(v4bf*)(o + i) = r;
}

// ---------------- Stage 1a: depthwise conv (3-tap) + mask, transpose (B,C,T)->(B,T,C) f32
__global__ void dwconv_tr_kernel(const float* __restrict__ x,
                                 const uint8_t* __restrict__ mask,   // (B,T)
                                 const float* __restrict__ cw,       // (C,3)
                                 float* __restrict__ yt) {           // (B,T,C)
  __shared__ float tile[32][33];
  int blk = blockIdx.x;
  int tilesPerBatch = (CC / 32) * (TT / 32);
  int b = blk / tilesPerBatch;
  int rem = blk % tilesPerBatch;
  int c0 = (rem / (TT / 32)) * 32;
  int t0 = (rem % (TT / 32)) * 32;
  int tid = threadIdx.x;
#pragma unroll
  for (int k = 0; k < 4; ++k) {
    int idx = tid + 256 * k;
    int tl = idx & 31, cl = idx >> 5;
    int c = c0 + cl, t = t0 + tl;
    const float* xc = x + ((size_t)b * CC + c) * TT;
    float xm = (t > 0) ? xc[t - 1] : 0.f;
    float x0 = xc[t];
    float xp = (t < TT - 1) ? xc[t + 1] : 0.f;
    float mv = mask[b * TT + t] ? 1.f : 0.f;
    tile[cl][tl] = (cw[c * 3 + 0] * xm + cw[c * 3 + 1] * x0 + cw[c * 3 + 2] * xp) * mv;
  }
  __syncthreads();
#pragma unroll
  for (int k = 0; k < 4; ++k) {
    int idx = tid + 256 * k;
    int cl = idx & 31, tl = idx >> 5;
    yt[((size_t)b * TT + t0 + tl) * CC + c0 + cl] = tile[cl][tl];
  }
}

// ---------------- Stage 1b: channel LayerNorm over contiguous (T,C) rows -> bf16 (B,T,C)
__global__ void ln_kernel(const float* __restrict__ yt,
                          const float* __restrict__ lnw,
                          const float* __restrict__ lnb,
                          bf16_t* __restrict__ out) {
  __shared__ float s_sum[256];
  __shared__ float s_sq[256];
  int tid = threadIdx.x;
  size_t rowoff = (size_t)blockIdx.x * CC;      // blockIdx = b*T + t
  float4 y = *(const float4*)(yt + rowoff + tid * 4);
  s_sum[tid] = y.x + y.y + y.z + y.w;
  s_sq[tid] = y.x * y.x + y.y * y.y + y.z * y.z + y.w * y.w;
  __syncthreads();
  for (int off = 128; off > 0; off >>= 1) {
    if (tid < off) { s_sum[tid] += s_sum[tid + off]; s_sq[tid] += s_sq[tid + off]; }
    __syncthreads();
  }
  float mu = s_sum[0] * (1.f / CC);
  float var = s_sq[0] * (1.f / CC) - mu * mu;
  float rstd = rsqrtf(var + LN_EPS);
  float4 w4 = *(const float4*)(lnw + tid * 4);
  float4 b4 = *(const float4*)(lnb + tid * 4);
  v4bf o;
  o[0] = (bf16_t)((y.x - mu) * rstd * w4.x + b4.x);
  o[1] = (bf16_t)((y.y - mu) * rstd * w4.y + b4.y);
  o[2] = (bf16_t)((y.z - mu) * rstd * w4.z + b4.z);
  o[3] = (bf16_t)((y.w - mu) * rstd * w4.w + b4.w);
  *(v4bf*)(out + rowoff + tid * 4) = o;
}

// ---------------- Stage 2/4: Out[b,o,t] = (sum_c W[o,c]*X[b,t,c] + bias[o]) * mask * oscale
// X bf16 (B,T,C), W bf16 row-major. OUT_TC: bf16 (B,T,C) b128 store; else (B,C,T) scalar.
template <bool MASK, bool OUT_TC, typename OutT>
__global__ void gemm1x1_wmma(const bf16_t* __restrict__ W,
                             const float* __restrict__ bias,
                             const bf16_t* __restrict__ X,
                             OutT* __restrict__ Out,
                             const uint8_t* __restrict__ mask,
                             float oscale) {
  int wave = blockIdx.x * (blockDim.x >> 5) + (threadIdx.x >> 5);
  int lane = threadIdx.x & 31;
  int half = lane >> 4;
  int l16 = lane & 15;
  int tilesPerBatch = (CC / 16) * (TT / 16);
  int b = wave / tilesPerBatch;
  int rem = wave % tilesPerBatch;
  int m0 = (rem / (TT / 16)) * 16;
  int n0 = (rem % (TT / 16)) * 16;
  const bf16_t* Wrow = W + (size_t)(m0 + l16) * CC + half * 16;           // A: K-run of 16
  const bf16_t* Xrow = X + ((size_t)b * TT + n0 + l16) * CC + half * 16;  // B: K-run of 16
  v8f acc = {};
  for (int k0 = 0; k0 < CC; k0 += 32) {
    v16bf a = *(const v16bf*)(Wrow + k0);
    v16bf bm = *(const v16bf*)(Xrow + k0);
    acc = __builtin_amdgcn_wmma_f32_16x16x32_bf16(false, a, false, bm,
                                                  (short)0, acc, false, false);
  }
  float mk = oscale;
  if (MASK) mk = mask[b * TT + n0 + l16] ? oscale : 0.f;
  const float4 blo = *(const float4*)(bias + m0 + 8 * half);
  const float4 bhi = *(const float4*)(bias + m0 + 8 * half + 4);
  float bb[8] = {blo.x, blo.y, blo.z, blo.w, bhi.x, bhi.y, bhi.z, bhi.w};
  if constexpr (OUT_TC) {
    v8bf o;
#pragma unroll
    for (int r = 0; r < 8; ++r) o[r] = (bf16_t)((acc[r] + bb[r]) * mk);
    *(v8bf*)((bf16_t*)Out + ((size_t)b * TT + n0 + l16) * CC + m0 + 8 * half) = o;
  } else {
#pragma unroll
    for (int r = 0; r < 8; ++r) {
      int m = m0 + r + 8 * half;
      Out[(size_t)b * CC * TT + (size_t)m * TT + n0 + l16] =
          (OutT)((acc[r] + bb[r]) * mk);
    }
  }
}

// ---------------- Stage 3: flash-style attention, one wave per (b,h,16-row S tile).
// Q/K bf16 (B,T,C); V bf16 (B,C,T) so each lane loads a B-tile as two contiguous v8bf runs along T.
// Softmax scale pre-folded into Q projection; kv-mask pre-folded into V projection.
__global__ void attn_wmma(const bf16_t* __restrict__ Q,
                          const bf16_t* __restrict__ K,
                          const bf16_t* __restrict__ V,
                          bf16_t* __restrict__ O) {
  __shared__ bf16_t plds[4][16 * 32];   // per-wave prob tile, C/D -> A transpose
  int wid  = threadIdx.x >> 5;
  int lane = threadIdx.x & 31;
  int half = lane >> 4;
  int l16  = lane & 15;
  int wave = blockIdx.x * 4 + wid;
  int bh = wave / (TT / 16);
  int st = wave % (TT / 16);
  int b = bh / HH, h = bh % HH;
  int s0 = st * 16;

  const bf16_t* Qrow = Q + ((size_t)b * TT + s0 + l16) * CC + h * CHD;
  v16bf aq0 = *(const v16bf*)(Qrow + half * 16);
  v16bf aq1 = *(const v16bf*)(Qrow + 32 + half * 16);

  const bf16_t* Kb = K + (size_t)b * TT * CC + h * CHD;
  const bf16_t* Vb = V + ((size_t)b * CC + h * CHD) * TT;   // (C,T) layout

  float row_max[8], row_sum[8];
  v8f acc[4];
  v8f vz = {};
#pragma unroll
  for (int r = 0; r < 8; ++r) { row_max[r] = -1e30f; row_sum[r] = 0.f; }
#pragma unroll
  for (int j = 0; j < 4; ++j) acc[j] = vz;

  for (int tc = 0; tc < TT; tc += 32) {
    const bf16_t* kr0 = Kb + (size_t)(tc + l16) * CC + half * 16;
    const bf16_t* kr1 = kr0 + (size_t)16 * CC;
    v8f sc0 = {}, sc1 = {};
    sc0 = __builtin_amdgcn_wmma_f32_16x16x32_bf16(false, aq0, false, *(const v16bf*)(kr0),
                                                  (short)0, sc0, false, false);
    sc0 = __builtin_amdgcn_wmma_f32_16x16x32_bf16(false, aq1, false, *(const v16bf*)(kr0 + 32),
                                                  (short)0, sc0, false, false);
    sc1 = __builtin_amdgcn_wmma_f32_16x16x32_bf16(false, aq0, false, *(const v16bf*)(kr1),
                                                  (short)0, sc1, false, false);
    sc1 = __builtin_amdgcn_wmma_f32_16x16x32_bf16(false, aq1, false, *(const v16bf*)(kr1 + 32),
                                                  (short)0, sc1, false, false);

    // online softmax across the 32 new columns (fused DPP16 row reductions)
#pragma unroll
    for (int r = 0; r < 8; ++r) {
      float a0 = sc0[r], a1 = sc1[r];       // scale already folded into Q
      float mnew = fmaxf(row_max[r], dpp_max16(fmaxf(a0, a1)));
      float corr = __expf(row_max[r] - mnew);
      row_max[r] = mnew;
      float p0 = __expf(a0 - mnew), p1 = __expf(a1 - mnew);
      row_sum[r] = row_sum[r] * corr + dpp_sum16(p0 + p1);
#pragma unroll
      for (int j = 0; j < 4; ++j) acc[j][r] *= corr;
      int m = r + 8 * half;
      plds[wid][m * 32 + l16]      = (bf16_t)p0;
      plds[wid][m * 32 + 16 + l16] = (bf16_t)p1;
    }
    // LDS region is per-wave private and same-wave LDS ops are in-order (DScnt);
    // only a compiler fence is needed to keep the punned vector load below the stores.
    asm volatile("" ::: "memory");
    const bf16_t* pr = &plds[wid][l16 * 32];
    v16bf ap = SHUF16(*(const v8bf*)(pr + half * 8),
                      *(const v8bf*)(pr + 16 + half * 8));
    asm volatile("" ::: "memory");

    // O += P * V : V (C,T) layout -> two K-runs of 8 per lane (k = 8h+s, 16+8h+s-8), two v8bf per tile
#pragma unroll
    for (int j = 0; j < 4; ++j) {
      v16bf bv = SHUF16(*(const v8bf*)(Vb + (size_t)(j * 16 + l16) * TT + tc + half * 8),
                        *(const v8bf*)(Vb + (size_t)(j * 16 + l16) * TT + tc + 16 + half * 8));
      acc[j] = __builtin_amdgcn_wmma_f32_16x16x32_bf16(false, ap, false, bv,
                                                       (short)0, acc[j], false, false);
    }
  }

  float rcp[8];
#pragma unroll
  for (int r = 0; r < 8; ++r) rcp[r] = 1.f / row_sum[r];
#pragma unroll
  for (int j = 0; j < 4; ++j) {
#pragma unroll
    for (int r = 0; r < 8; ++r) {
      O[((size_t)b * TT + s0 + r + 8 * half) * CC + h * CHD + j * 16 + l16] =
          (bf16_t)(acc[j][r] * rcp[r]);
    }
  }
}

// second tuple output: query_mask as float
__global__ void write_mask_kernel(const uint8_t* __restrict__ qm,
                                  float* __restrict__ out, int n) {
  int i = blockIdx.x * blockDim.x + threadIdx.x;
  if (i < n) out[i] = qm[i] ? 1.f : 0.f;
}

extern "C" void kernel_launch(void* const* d_in, const int* in_sizes, int n_in,
                              void* d_out, int out_size, void* d_ws, size_t ws_size,
                              hipStream_t stream) {
  (void)in_sizes; (void)n_in; (void)ws_size;
  const float*   query = (const float*)d_in[0];
  const float*   key   = (const float*)d_in[1];
  const float*   value = (const float*)d_in[2];
  const uint8_t* qm    = (const uint8_t*)d_in[3];
  const uint8_t* kvm   = (const uint8_t*)d_in[4];
  const float* qcw = (const float*)d_in[5];
  const float* kcw = (const float*)d_in[6];
  const float* vcw = (const float*)d_in[7];
  const float* qnw = (const float*)d_in[8];
  const float* qnb = (const float*)d_in[9];
  const float* knw = (const float*)d_in[10];
  const float* knb = (const float*)d_in[11];
  const float* vnw = (const float*)d_in[12];
  const float* vnb = (const float*)d_in[13];
  const float* wq = (const float*)d_in[14];
  const float* bq = (const float*)d_in[15];
  const float* wk = (const float*)d_in[16];
  const float* bk = (const float*)d_in[17];
  const float* wv = (const float*)d_in[18];
  const float* bv = (const float*)d_in[19];
  const float* wp = (const float*)d_in[20];
  const float* bp = (const float*)d_in[21];

  const size_t SZ = (size_t)BB * CC * TT;
  const size_t WSZ = (size_t)CC * CC;
  float*  yt  = (float*)d_ws;                 // 16 MB, reused across q/k/v
  bf16_t* xnq = (bf16_t*)(yt + SZ);
  bf16_t* xnk = xnq + SZ;
  bf16_t* xnv = xnk + SZ;
  bf16_t* pq  = xnv + SZ;
  bf16_t* pk  = pq + SZ;
  bf16_t* pv  = pk + SZ;                      // (B,C,T) layout
  bf16_t* ao  = pv + SZ;
  bf16_t* wqb = ao + SZ;
  bf16_t* wkb = wqb + WSZ;
  bf16_t* wvb = wkb + WSZ;
  bf16_t* wpb = wvb + WSZ;

  // Stage 0: weights -> bf16 (L2-resident afterwards)
  const int cvtBlocks = (int)(WSZ / 4 / 256);
  cvt_w_kernel<<<cvtBlocks, 256, 0, stream>>>(wq, wqb);
  cvt_w_kernel<<<cvtBlocks, 256, 0, stream>>>(wk, wkb);
  cvt_w_kernel<<<cvtBlocks, 256, 0, stream>>>(wv, wvb);
  cvt_w_kernel<<<cvtBlocks, 256, 0, stream>>>(wp, wpb);

  const int trBlocks = BB * (CC / 32) * (TT / 32);
  const int lnBlocks = BB * TT;
  // Stage 1: conv+mask+transpose, then LN (sequential reuse of yt)
  dwconv_tr_kernel<<<trBlocks, 256, 0, stream>>>(query, qm, qcw, yt);
  ln_kernel<<<lnBlocks, 256, 0, stream>>>(yt, qnw, qnb, xnq);
  dwconv_tr_kernel<<<trBlocks, 256, 0, stream>>>(key, kvm, kcw, yt);
  ln_kernel<<<lnBlocks, 256, 0, stream>>>(yt, knw, knb, xnk);
  dwconv_tr_kernel<<<trBlocks, 256, 0, stream>>>(value, kvm, vcw, yt);
  ln_kernel<<<lnBlocks, 256, 0, stream>>>(yt, vnw, vnb, xnv);

  // Stage 2: projections. Q carries the 1/sqrt(ch)=0.125 softmax scale;
  // V carries the post-bias kv mask and is written in (B,C,T) for contiguous B-tiles.
  const int gemm_blocks = (BB * (CC / 16) * (TT / 16)) / 8;
  gemm1x1_wmma<false, true, bf16_t><<<gemm_blocks, 256, 0, stream>>>(wqb, bq, xnq, pq, nullptr, 0.125f);
  gemm1x1_wmma<false, true, bf16_t><<<gemm_blocks, 256, 0, stream>>>(wkb, bk, xnk, pk, nullptr, 1.0f);
  gemm1x1_wmma<true, false, bf16_t><<<gemm_blocks, 256, 0, stream>>>(wvb, bv, xnv, pv, kvm, 1.0f);

  // Stage 3: fused online-softmax attention
  attn_wmma<<<(BB * HH * (TT / 16)) / 4, 128, 0, stream>>>(pq, pk, pv, ao);

  // Stage 4: output projection + bias + query mask -> f32 (B,C,T) d_out
  gemm1x1_wmma<true, false, float><<<gemm_blocks, 256, 0, stream>>>(wpb, bp, ao, (float*)d_out, qm, 1.0f);

  // Stage 5: second tuple output (query_mask) if space reserved
  const int nmask = BB * TT;
  if (out_size >= (int)(SZ + (size_t)nmask)) {
    write_mask_kernel<<<(nmask + 255) / 256, 256, 0, stream>>>(qm, (float*)d_out + SZ, nmask);
  }
}
